// IDKN_cat_10531259809802
// MI455X (gfx1250) — matrix operand order repaired and checked by reading, not compiled, not run on hardware
//
#include <hip/hip_runtime.h>
#include <hip/hip_bf16.h>

#define NNODES 10000
#define NEDGES 320000
#define ETOT   (NEDGES + NNODES)
#define MASK_W 313            // ceil(10000/32) u32 words per adjacency row

typedef __attribute__((ext_vector_type(16))) _Float16 v16h;
typedef __attribute__((ext_vector_type(8)))  float    v8f;

__device__ __forceinline__ float lrelu(float x) { return x >= 0.0f ? x : 0.2f * x; }

// order-preserving float <-> uint mapping for atomicMax-based segment max
__device__ __forceinline__ unsigned fflip(float f) {
    unsigned u = __float_as_uint(f);
    return (u & 0x80000000u) ? ~u : (u | 0x80000000u);
}
__device__ __forceinline__ float funflip(unsigned u) {
    return __uint_as_float((u & 0x80000000u) ? (u ^ 0x80000000u) : ~u);
}

// ---------------------------------------------------------------- utilities
__global__ void zero_kernel(unsigned* __restrict__ p, int n) {
    int i = blockIdx.x * blockDim.x + threadIdx.x;
    if (i < n) p[i] = 0u;
}

__global__ void concat_kernel(const float* __restrict__ x1, const float* __restrict__ x2,
                              float* __restrict__ xcat) {
    int i = blockIdx.x * blockDim.x + threadIdx.x;
    if (i >= NNODES * 20) return;
    int n = i / 20, c = i % 20;
    xcat[i] = (c < 10) ? x1[n * 10 + c] : x2[n * 10 + (c - 10)];
}

// ------------------------------------------------- WMMA feature transform
// Hout[N, HD] = X[N, K] @ W[K, HD], K <= 32 (zero padded), one wave per 16x16 tile.
// Fully static bounds: lane-dependent OOB handled via clamped index + select
// (no exec-masked per-element loads). Operand packing per CDNA5 ISA 7.12.2.
template <int K, int HD, int TILESN>
__global__ void gat_gemm_wmma(const float* __restrict__ X, const float* __restrict__ W,
                              float* __restrict__ Hout) {
    int tile = blockIdx.x;
    int tm = tile / TILESN;
    int tn = tile % TILESN;
    int lane = threadIdx.x & 31;
    int r  = lane & 15;
    int hi = lane >> 4;

    int rowA = tm * 16 + r;          // M index held by this lane (A operand)
    int colB = tn * 16 + r;          // N index held by this lane (B operand)
    int colBc = (colB < HD) ? colB : 0;
    const float* xrow = X + (size_t)rowA * K;

    v16h a, b;
#pragma unroll
    for (int e = 0; e < 16; ++e) {
        const int v = e >> 1, t = e & 1;
        // A (16x32 MxK): lanes 0-15 -> K 0..7 (v0..3), 16..23 (v4..7); lanes 16-31 -> +8
        int ka = (v < 4) ? (2 * v + t + 8 * hi) : (16 + 2 * (v - 4) + t + 8 * hi);
        // B (32x16 KxN): same K order as A: lanes 0-15 -> K 0..7, 16..23; lanes 16-31 -> +8
        int kb = (v < 4) ? (2 * v + t + 8 * hi) : (16 + 2 * (v - 4) + t + 8 * hi);
        int kac = (ka < K) ? ka : 0;                 // clamp -> always in-bounds load
        int kbc = (kb < K) ? kb : 0;
        float av = xrow[kac];
        float bv = W[kbc * HD + colBc];
        av = (ka < K) ? av : 0.0f;                   // v_cndmask, no branch
        bv = (kb < K && colB < HD) ? bv : 0.0f;
        a[e] = (_Float16)av;
        b[e] = (_Float16)bv;
    }
    v8f c = {};
    c = __builtin_amdgcn_wmma_f32_16x16x32_f16(false, a, false, b, (short)0, c, false, false);
#pragma unroll
    for (int g = 0; g < 8; ++g) {
        int m   = tm * 16 + g + 8 * hi;   // C/D: VGPR g holds M=g (lanes<16) / g+8 (lanes>=16)
        int col = tn * 16 + r;
        if (col < HD) Hout[(size_t)m * HD + col] = c[g];
    }
}

// ------------------------------------------------- per-node attention logits
template <int H>
__global__ void alpha_kernel(const float* __restrict__ h, const float* __restrict__ a_s,
                             const float* __restrict__ a_d, float* __restrict__ as_o,
                             float* __restrict__ ad_o) {
    int i = blockIdx.x * blockDim.x + threadIdx.x;
    if (i >= NNODES * H) return;
    int n = i / H, hh = i % H;
    const float4 hv = ((const float4*)h)[n * H + hh];
    const float4 sv = ((const float4*)a_s)[hh];
    const float4 dv = ((const float4*)a_d)[hh];
    as_o[i] = hv.x * sv.x + hv.y * sv.y + hv.z * sv.z + hv.w * sv.w;
    ad_o[i] = hv.x * dv.x + hv.y * dv.y + hv.z * dv.z + hv.w * dv.w;
}

// ------------------------------------------------- edge pass 1: segment max
template <int H>
__global__ void edge_max_kernel(const int* __restrict__ ei, const float* __restrict__ as_b,
                                const float* __restrict__ ad_b, unsigned* __restrict__ emax) {
    int e = blockIdx.x * blockDim.x + threadIdx.x;
    if (e >= ETOT) return;
    int s, d;
    if (e < NEDGES) { s = ei[e]; d = ei[NEDGES + e]; } else { s = d = e - NEDGES; }
#pragma unroll
    for (int h = 0; h < H; ++h) {
        float ev = lrelu(as_b[s * H + h] + ad_b[d * H + h]);
        atomicMax(&emax[d * H + h], fflip(ev));
    }
}

// ------------------- edge pass 2: exp-weights, denom + weighted aggregation
template <int H>
__global__ void edge_sum_kernel(const int* __restrict__ ei, const float* __restrict__ as_b,
                                const float* __restrict__ ad_b, const unsigned* __restrict__ emax,
                                float* __restrict__ esum, const float* __restrict__ hfeat,
                                float* __restrict__ accum) {
    int e = blockIdx.x * blockDim.x + threadIdx.x;
    if (e >= ETOT) return;
    int s, d;
    if (e < NEDGES) { s = ei[e]; d = ei[NEDGES + e]; } else { s = d = e - NEDGES; }
    const float4* h4 = (const float4*)hfeat;
#pragma unroll
    for (int h = 0; h < H; ++h) {
        float ev = lrelu(as_b[s * H + h] + ad_b[d * H + h]);
        float m  = funflip(emax[d * H + h]);
        float w  = __expf(ev - m);
        atomicAdd(&esum[d * H + h], w);
        float4 hv = h4[s * H + h];                    // 16B gather of h[src, head, :]
        float* ap = &accum[(d * H + h) * 4];
        atomicAdd(ap + 0, w * hv.x);
        atomicAdd(ap + 1, w * hv.y);
        atomicAdd(ap + 2, w * hv.z);
        atomicAdd(ap + 3, w * hv.w);
    }
}

// ------------------------------------------------- normalize + bias (in place)
template <int H>
__global__ void finish_kernel(float* __restrict__ accum, const float* __restrict__ esum,
                              const float* __restrict__ bias) {
    constexpr int HD = H * 4;
    int i = blockIdx.x * blockDim.x + threadIdx.x;
    if (i >= NNODES * HD) return;
    int n = i / HD, c = i % HD;
    accum[i] = accum[i] / esum[n * H + (c >> 2)] + bias[c];
}

// -------------------------------- dedup adjacency bitmask (row = src node)
__global__ void mask_kernel(const int* __restrict__ ei, unsigned* __restrict__ mask) {
    int e = blockIdx.x * blockDim.x + threadIdx.x;
    if (e >= ETOT) return;
    int s, d;
    if (e < NEDGES) { s = ei[e]; d = ei[NEDGES + e]; } else { s = d = e - NEDGES; }
    atomicOr(&mask[s * MASK_W + (d >> 5)], 1u << (d & 31));
}

// ---------------- final scoring: one wave32 per row of the dedup'd adjacency
// out[i] = (sum_j dot(x6_i,x6_j))/deg_i + x1[i,0] + dot(sum_j x6_j, lin2)/deg_i
__global__ void row_scan_kernel(const unsigned* __restrict__ mask, const float* __restrict__ x6,
                                const float* __restrict__ x1, const float* __restrict__ lin2,
                                float* __restrict__ out) {
    int wave = (blockIdx.x * blockDim.x + threadIdx.x) >> 5;
    int lane = threadIdx.x & 31;
    if (wave >= NNODES) return;
    int row = wave;

    const float4* x64 = (const float4*)x6;
    float4 xiA = x64[row * 2 + 0];
    float4 xiB = x64[row * 2 + 1];

    float4 accA = {0, 0, 0, 0}, accB = {0, 0, 0, 0};
    float dotacc = 0.0f;
    int   cnt = 0;

    for (int w = lane; w < MASK_W; w += 32) {
        unsigned bits = mask[row * MASK_W + w];
        while (bits) {
            int b = __ffs(bits) - 1;
            bits &= bits - 1;
            int j = (w << 5) + b;
            ++cnt;
            float4 vA = x64[j * 2 + 0];
            float4 vB = x64[j * 2 + 1];
            accA.x += vA.x; accA.y += vA.y; accA.z += vA.z; accA.w += vA.w;
            accB.x += vB.x; accB.y += vB.y; accB.z += vB.z; accB.w += vB.w;
            dotacc += xiA.x * vA.x + xiA.y * vA.y + xiA.z * vA.z + xiA.w * vA.w
                    + xiB.x * vB.x + xiB.y * vB.y + xiB.z * vB.z + xiB.w * vB.w;
        }
    }
    // wave32 butterfly reduction
    float a0 = accA.x, a1 = accA.y, a2 = accA.z, a3 = accA.w;
    float a4 = accB.x, a5 = accB.y, a6 = accB.z, a7 = accB.w;
#pragma unroll
    for (int off = 16; off >= 1; off >>= 1) {
        dotacc += __shfl_xor(dotacc, off, 32);
        cnt    += __shfl_xor(cnt, off, 32);
        a0 += __shfl_xor(a0, off, 32); a1 += __shfl_xor(a1, off, 32);
        a2 += __shfl_xor(a2, off, 32); a3 += __shfl_xor(a3, off, 32);
        a4 += __shfl_xor(a4, off, 32); a5 += __shfl_xor(a5, off, 32);
        a6 += __shfl_xor(a6, off, 32); a7 += __shfl_xor(a7, off, 32);
    }
    if (lane == 0) {
        float inv = 1.0f / (float)cnt;      // self-loop guarantees cnt >= 1
        float g = a0 * lin2[0] + a1 * lin2[1] + a2 * lin2[2] + a3 * lin2[3]
                + a4 * lin2[4] + a5 * lin2[5] + a6 * lin2[6] + a7 * lin2[7];
        out[row] = dotacc * inv + x1[row * 10] + g * inv;
    }
}

// ------------------------------------------------------------------ driver
template <int K, int H>
static void run_gat_layer(const float* xin,
                          const float* W, const float* a_s, const float* a_d, const float* bias,
                          float* h_buf, float* accum, float* as_b, float* ad_b,
                          unsigned* emax, float* esum, const int* ei, hipStream_t stream) {
    constexpr int HD = H * 4;
    constexpr int TILESN = (HD + 15) / 16;
    gat_gemm_wmma<K, HD, TILESN>
        <<<dim3((NNODES / 16) * TILESN), dim3(32), 0, stream>>>(xin, W, h_buf);
    alpha_kernel<H><<<dim3((NNODES * H + 255) / 256), dim3(256), 0, stream>>>(h_buf, a_s, a_d, as_b, ad_b);
    zero_kernel<<<dim3((NNODES * H + 255) / 256), dim3(256), 0, stream>>>(emax, NNODES * H);
    zero_kernel<<<dim3((NNODES * H + 255) / 256), dim3(256), 0, stream>>>((unsigned*)esum, NNODES * H);
    zero_kernel<<<dim3((NNODES * HD + 255) / 256), dim3(256), 0, stream>>>((unsigned*)accum, NNODES * HD);
    int eb = (ETOT + 255) / 256;
    edge_max_kernel<H><<<dim3(eb), dim3(256), 0, stream>>>(ei, as_b, ad_b, emax);
    edge_sum_kernel<H><<<dim3(eb), dim3(256), 0, stream>>>(ei, as_b, ad_b, emax, esum, h_buf, accum);
    finish_kernel<H><<<dim3((NNODES * HD + 255) / 256), dim3(256), 0, stream>>>(accum, esum, bias);
}

extern "C" void kernel_launch(void* const* d_in, const int* in_sizes, int n_in,
                              void* d_out, int out_size, void* d_ws, size_t ws_size,
                              hipStream_t stream) {
    const float* x1  = (const float*)d_in[0];
    const float* x2  = (const float*)d_in[1];
    const int*   ei  = (const int*)d_in[2];
    const float* W1  = (const float*)d_in[4];
    const float* as1 = (const float*)d_in[5];
    const float* ad1 = (const float*)d_in[6];
    const float* b1  = (const float*)d_in[7];
    const float* W2  = (const float*)d_in[8];
    const float* as2 = (const float*)d_in[9];
    const float* ad2 = (const float*)d_in[10];
    const float* b2  = (const float*)d_in[11];
    const float* W3  = (const float*)d_in[12];
    const float* as3 = (const float*)d_in[13];
    const float* ad3 = (const float*)d_in[14];
    const float* b3  = (const float*)d_in[15];
    const float* lin2 = (const float*)d_in[16];
    float* out = (float*)d_out;

    // workspace carve-up (all 256B-aligned sizes)
    char* ws = (char*)d_ws;
    size_t off = 0;
    float* xcat = (float*)(ws + off); off += (size_t)NNODES * 20 * 4;   // 800,000
    float* hbuf = (float*)(ws + off); off += (size_t)NNODES * 32 * 4;   // 1,280,000
    float* out1 = (float*)(ws + off); off += (size_t)NNODES * 32 * 4;
    float* out2 = (float*)(ws + off); off += (size_t)NNODES * 32 * 4;
    float* out3 = (float*)(ws + off); off += (size_t)NNODES * 32 * 4;
    float* as_b = (float*)(ws + off); off += (size_t)NNODES * 8 * 4;
    float* ad_b = (float*)(ws + off); off += (size_t)NNODES * 8 * 4;
    unsigned* emax = (unsigned*)(ws + off); off += (size_t)NNODES * 8 * 4;
    float* esum = (float*)(ws + off); off += (size_t)NNODES * 8 * 4;
    unsigned* mask = (unsigned*)(ws + off); off += (size_t)NNODES * MASK_W * 4;  // 12,520,000

    // x = concat(x1, x2)
    concat_kernel<<<dim3((NNODES * 20 + 255) / 256), dim3(256), 0, stream>>>(x1, x2, xcat);

    // three GAT layers: (20 -> 8x4), (32 -> 4x4), (16 -> 2x4)
    run_gat_layer<20, 8>(xcat, W1, as1, ad1, b1, hbuf, out1, as_b, ad_b, emax, esum, ei, stream);
    run_gat_layer<32, 4>(out1, W2, as2, ad2, b2, hbuf, out2, as_b, ad_b, emax, esum, ei, stream);
    run_gat_layer<16, 2>(out2, W3, as3, ad3, b3, hbuf, out3, as_b, ad_b, emax, esum, ei, stream);

    // deduplicated adjacency bitmask (matches .set(1.0) semantics + diagonal)
    zero_kernel<<<dim3((NNODES * MASK_W + 255) / 256), dim3(256), 0, stream>>>(mask, NNODES * MASK_W);
    mask_kernel<<<dim3((ETOT + 255) / 256), dim3(256), 0, stream>>>(ei, mask);

    // final score: R2 + x1[:,0] + global_score, one wave per row
    row_scan_kernel<<<dim3((NNODES * 32 + 255) / 256), dim3(256), 0, stream>>>(mask, out3, x1, lin2, out);
}
